// CrossAttention_84078279786935
// MI455X (gfx1250) — matrix operand order repaired and checked by reading, not compiled, not run on hardware
//
#include <hip/hip_runtime.h>
#include <hip/hip_bf16.h>

typedef __bf16 bf16_t;
typedef __attribute__((ext_vector_type(16))) __bf16 v16bf;
typedef __attribute__((ext_vector_type(8)))  __bf16 v8bf;
typedef __attribute__((ext_vector_type(8)))  float  v8f;

#define B_    16
#define N_    4096
#define M_    77
#define MPAD  96      // padded M for PV K-dim (multiple of 32)
#define MPAD2 80      // padded M for S col tiles (multiple of 16)
#define DQ    320
#define DC    768
#define INNER 320
#define HEADS 8
#define DHEAD 40
#define DPAD  64      // padded head dim for S K-dim (multiple of 32)
#define DPAD2 48      // padded head dim for PV col tiles (multiple of 16)
#define SCALE_F 0.15811388300841897f  // 40^-0.5

// ---------- fragment load helpers (ISA bf16 WMMA layouts, wave32) ----------

// B fragment: two 8-element chunks at +0 and +16 (lane = column; caller adds lane-half K base +0/+8)
__device__ __forceinline__ v16bf load_contig16(const bf16_t* __restrict__ p) {
    v8bf lo = *(const v8bf*)p;
    v8bf hi = *(const v8bf*)(p + 16);
    v16bf r;
#pragma unroll
    for (int i = 0; i < 8; ++i) { r[i] = lo[i]; r[i + 8] = hi[i]; }
    return r;
}

// A fragment: two 8-element chunks at +0 and +16 (caller adds lane-half K base +0/+8)
__device__ __forceinline__ v16bf load_afrag(const bf16_t* __restrict__ p) {
    v8bf lo = *(const v8bf*)p;
    v8bf hi = *(const v8bf*)(p + 16);
    v16bf r;
#pragma unroll
    for (int i = 0; i < 8; ++i) { r[i] = lo[i]; r[i + 8] = hi[i]; }
    return r;
}

// ---------- elementwise / layout prep kernels ----------

__global__ void cvt_f32_bf16(const float* __restrict__ s, bf16_t* __restrict__ d, int n) {
    int i = blockIdx.x * blockDim.x + threadIdx.x;
    if (i < n) d[i] = (bf16_t)s[i];
}

// W[K][N] (f32) -> WT[N][K] (bf16)
__global__ void transpose_bf16(const float* __restrict__ W, bf16_t* __restrict__ WT, int K, int N) {
    int idx = blockIdx.x * blockDim.x + threadIdx.x;
    if (idx < K * N) {
        int k = idx / N, j = idx % N;
        WT[(size_t)j * K + k] = (bf16_t)W[idx];
    }
}

__global__ void zero_bf16(bf16_t* __restrict__ d, int n) {
    int i = blockIdx.x * blockDim.x + threadIdx.x;
    if (i < n) d[i] = (bf16_t)0.0f;
}

// zero only the head-dim padding of qp [rows][HEADS][DPAD]
__global__ void zero_qpad(bf16_t* __restrict__ qp, int rows) {
    const int padw = DPAD - DHEAD;            // 24
    int idx = blockIdx.x * blockDim.x + threadIdx.x;
    int total = rows * HEADS * padw;
    if (idx < total) {
        int per = HEADS * padw;
        int r = idx / per, rem = idx % per;
        int h = rem / padw, jj = DHEAD + rem % padw;
        qp[((size_t)r * HEADS + h) * DPAD + jj] = (bf16_t)0.0f;
    }
}

// ---------- generic bf16 WMMA GEMM: C = A[MxK] * BT[NxK]^T ----------
// one wave computes a 16x64 strip of C (4 WMMA tiles sharing the A fragment).
// MODE 0: f32 row-major + bias (final projection)
// MODE 1: qp layout [row][h][DPAD]        (q projection, d-padded)
// MODE 2: kp layout [b][h][MPAD2][DPAD]   (k projection, M- and d-padded)
// MODE 3: vp layout [b][h][DPAD2][MPAD]   (v projection, transposed + padded)
template <int MODE>
__global__ __launch_bounds__(256) void gemm_bf16(const bf16_t* __restrict__ A,
                                                 const bf16_t* __restrict__ BT,
                                                 void* __restrict__ Cout,
                                                 const float* __restrict__ bias,
                                                 int M, int N, int K) {
    const int lane = threadIdx.x & 31;
    const int wave = blockIdx.x * 8 + (threadIdx.x >> 5);
    const int strips = N >> 6;
    const int total = (M >> 4) * strips;
    if (wave >= total) return;                 // wave-uniform: EXEC stays all-1s for WMMA
    const int rowTile = wave / strips;
    const int strip   = wave % strips;
    const int half    = lane >> 4;             // 0/1
    const int l16     = lane & 15;

    v8f acc[4];
#pragma unroll
    for (int t = 0; t < 4; ++t)
#pragma unroll
        for (int i = 0; i < 8; ++i) acc[t][i] = 0.0f;

    const bf16_t* arow  = A  + (size_t)(rowTile * 16 + l16) * K + half * 8;
    const bf16_t* bcol0 = BT + (size_t)(strip * 64 + l16) * K + half * 8;

    for (int k = 0; k < K; k += 32) {
        if (k + 32 < K) __builtin_prefetch(arow + k + 32, 0, 1);  // global_prefetch_b8
        v16bf a = load_afrag(arow + k);
#pragma unroll
        for (int t = 0; t < 4; ++t) {
            v16bf b = load_contig16(bcol0 + (size_t)t * 16 * K + k);
            acc[t] = __builtin_amdgcn_wmma_f32_16x16x32_bf16(false, a, false, b,
                                                             (short)0, acc[t], false, false);
        }
    }

    const int r0 = rowTile * 16 + half * 8;    // C layout: VGPR i -> row r0+i, lane -> col
#pragma unroll
    for (int t = 0; t < 4; ++t) {
        const int c = strip * 64 + t * 16 + l16;
        const int h = c / DHEAD, jj = c % DHEAD;
#pragma unroll
        for (int i = 0; i < 8; ++i) {
            const int r = r0 + i;
            if (MODE == 0) {
                ((float*)Cout)[(size_t)r * N + c] = acc[t][i] + bias[c];
            } else if (MODE == 1) {
                ((bf16_t*)Cout)[((size_t)r * HEADS + h) * DPAD + jj] = (bf16_t)acc[t][i];
            } else if (MODE == 2) {
                int bb = r / M_, m = r % M_;
                ((bf16_t*)Cout)[((size_t)((bb * HEADS + h) * MPAD2 + m)) * DPAD + jj] =
                    (bf16_t)acc[t][i];
            } else {
                int bb = r / M_, m = r % M_;
                ((bf16_t*)Cout)[((size_t)((bb * HEADS + h) * DPAD2 + jj)) * MPAD + m] =
                    (bf16_t)acc[t][i];
            }
        }
    }
}

// ---------- fused attention: one wave per (b, h, 16-row tile of N) ----------
// S = (q k^T) * scale  ->  softmax over 77 cols  ->  O = P v
// All operands pre-padded in memory: every fragment load is unconditional.
__global__ __launch_bounds__(32) void attn_kernel(const bf16_t* __restrict__ qp,
                                                  const bf16_t* __restrict__ kp,
                                                  const bf16_t* __restrict__ vp,
                                                  bf16_t* __restrict__ obuf) {
    __shared__ __align__(16) float  S[16][MPAD2];  // 16 x 80 scores
    __shared__ __align__(16) bf16_t P[16][MPAD];   // 16 x 96 probs (zero padded)

    const int lane = threadIdx.x;
    const int wid  = blockIdx.x;
    const int nt   = wid & 255;
    const int h    = (wid >> 8) & 7;
    const int b    = wid >> 11;
    const int half = lane >> 4;
    const int l16  = lane & 15;

    // ---- S = q k^T  (K = DPAD = 64, 2 K-steps; 5 col tiles of 16 over MPAD2=80) ----
    const bf16_t* qrow = qp + (((size_t)(b * N_ + nt * 16 + l16)) * HEADS + h) * DPAD + half * 8;
    const bf16_t* kbh  = kp + ((size_t)(b * HEADS + h)) * MPAD2 * DPAD;
#pragma unroll
    for (int mt = 0; mt < 5; ++mt) {
        v8f acc;
#pragma unroll
        for (int i = 0; i < 8; ++i) acc[i] = 0.0f;
#pragma unroll
        for (int ks = 0; ks < 2; ++ks) {
            v16bf a = load_afrag(qrow + ks * 32);
            v16bf bfrag = load_contig16(kbh + (size_t)(mt * 16 + l16) * DPAD +
                                        ks * 32 + half * 8);
            acc = __builtin_amdgcn_wmma_f32_16x16x32_bf16(false, a, false, bfrag,
                                                          (short)0, acc, false, false);
        }
#pragma unroll
        for (int i = 0; i < 8; ++i)
            S[i + half * 8][mt * 16 + l16] = acc[i] * SCALE_F;
    }
    __syncthreads();

    // ---- softmax rows (lanes 0..15, one row each; cols 77..95 of P zero) ----
    if (lane < 16) {
        float mx = -1e30f;
        for (int m = 0; m < M_; ++m) mx = fmaxf(mx, S[lane][m]);
        float sum = 0.0f;
        for (int m = 0; m < M_; ++m) {
            float e = __expf(S[lane][m] - mx);
            S[lane][m] = e;
            sum += e;
        }
        float inv = 1.0f / sum;
        for (int m = 0; m < MPAD; ++m)
            P[lane][m] = (m < M_) ? (bf16_t)(S[lane][m] * inv) : (bf16_t)0.0f;
    }
    __syncthreads();

    // ---- O = P @ V  (K = MPAD = 96, 3 K-steps; 3 col tiles of 16 over DPAD2=48) ----
    const bf16_t* vbh  = vp + ((size_t)(b * HEADS + h)) * DPAD2 * MPAD;
    const bf16_t* prow = &P[l16][0];
#pragma unroll
    for (int jt = 0; jt < 3; ++jt) {
        v8f acc;
#pragma unroll
        for (int i = 0; i < 8; ++i) acc[i] = 0.0f;
        const int jj = jt * 16 + l16;
#pragma unroll
        for (int ks = 0; ks < 3; ++ks) {
            // A fragment from P (LDS): two aligned 16B ds loads
            const int kb0 = ks * 32 + half * 8;
            v8bf plo = *(const v8bf*)(prow + kb0);
            v8bf phi = *(const v8bf*)(prow + kb0 + 16);
            v16bf a;
#pragma unroll
            for (int i = 0; i < 8; ++i) { a[i] = plo[i]; a[i + 8] = phi[i]; }
            // B fragment from padded v^T
            v16bf bfrag = load_contig16(vbh + (size_t)jj * MPAD + ks * 32 + half * 8);
            acc = __builtin_amdgcn_wmma_f32_16x16x32_bf16(false, a, false, bfrag,
                                                          (short)0, acc, false, false);
        }
        if (jj < DHEAD) {
#pragma unroll
            for (int i = 0; i < 8; ++i)
                obuf[((size_t)(b * N_ + nt * 16 + i + half * 8)) * INNER + h * DHEAD + jj] =
                    (bf16_t)acc[i];
        }
    }
}

// ---------- host-side launch ----------

extern "C" void kernel_launch(void* const* d_in, const int* in_sizes, int n_in,
                              void* d_out, int out_size, void* d_ws, size_t ws_size,
                              hipStream_t stream) {
    const float* x   = (const float*)d_in[0];
    const float* ctx = (const float*)d_in[1];
    const float* Wq  = (const float*)d_in[2];
    const float* Wk  = (const float*)d_in[3];
    const float* Wv  = (const float*)d_in[4];
    const float* Wo  = (const float*)d_in[5];
    const float* bo  = (const float*)d_in[6];
    float* out = (float*)d_out;

    char* ws = (char*)d_ws;
    size_t off = 0;
    auto alloc = [&](size_t bytes) -> char* {
        char* p = ws + off;
        off += (bytes + 255) & ~(size_t)255;
        return p;
    };

    bf16_t* xb   = (bf16_t*)alloc((size_t)B_ * N_ * DQ * sizeof(bf16_t));
    bf16_t* cb   = (bf16_t*)alloc((size_t)B_ * M_ * DC * sizeof(bf16_t));
    bf16_t* WqT  = (bf16_t*)alloc((size_t)INNER * DQ * sizeof(bf16_t));
    bf16_t* WkT  = (bf16_t*)alloc((size_t)INNER * DC * sizeof(bf16_t));
    bf16_t* WvT  = (bf16_t*)alloc((size_t)INNER * DC * sizeof(bf16_t));
    bf16_t* WoT  = (bf16_t*)alloc((size_t)DQ * INNER * sizeof(bf16_t));
    bf16_t* qp   = (bf16_t*)alloc((size_t)B_ * N_ * HEADS * DPAD * sizeof(bf16_t));
    bf16_t* kp   = (bf16_t*)alloc((size_t)B_ * HEADS * MPAD2 * DPAD * sizeof(bf16_t));
    bf16_t* vp   = (bf16_t*)alloc((size_t)B_ * HEADS * DPAD2 * MPAD * sizeof(bf16_t));
    bf16_t* obuf = (bf16_t*)alloc((size_t)B_ * N_ * INNER * sizeof(bf16_t));

    const int T = 256;
    // converts
    {
        int n = B_ * N_ * DQ;
        cvt_f32_bf16<<<(n + T - 1) / T, T, 0, stream>>>(x, xb, n);
    }
    {
        int n = B_ * M_ * DC;
        cvt_f32_bf16<<<(n + T - 1) / T, T, 0, stream>>>(ctx, cb, n);
    }
    {
        int n = DQ * INNER;
        transpose_bf16<<<(n + T - 1) / T, T, 0, stream>>>(Wq, WqT, DQ, INNER);
        transpose_bf16<<<(n + T - 1) / T, T, 0, stream>>>(Wo, WoT, INNER, DQ);
    }
    {
        int n = DC * INNER;
        transpose_bf16<<<(n + T - 1) / T, T, 0, stream>>>(Wk, WkT, DC, INNER);
        transpose_bf16<<<(n + T - 1) / T, T, 0, stream>>>(Wv, WvT, DC, INNER);
    }
    // zero padded regions / buffers
    {
        int n = B_ * N_ * HEADS * (DPAD - DHEAD);
        zero_qpad<<<(n + T - 1) / T, T, 0, stream>>>(qp, B_ * N_);
    }
    {
        int n = B_ * HEADS * MPAD2 * DPAD;
        zero_bf16<<<(n + T - 1) / T, T, 0, stream>>>(kp, n);
    }
    {
        int n = B_ * HEADS * DPAD2 * MPAD;
        zero_bf16<<<(n + T - 1) / T, T, 0, stream>>>(vp, n);
    }

    // q projection: [65536 x 320] @ Wq -> qp (d-padded per head)
    {
        int M = B_ * N_, N = INNER, K = DQ;
        int waves = (M >> 4) * (N >> 6);
        gemm_bf16<1><<<(waves + 7) / 8, 256, 0, stream>>>(xb, WqT, (void*)qp, nullptr, M, N, K);
    }
    // k / v projections: [1232 x 768] @ Wk/Wv with padded per-head layouts
    {
        int M = B_ * M_, N = INNER, K = DC;
        int waves = (M >> 4) * (N >> 6);
        gemm_bf16<2><<<(waves + 7) / 8, 256, 0, stream>>>(cb, WkT, (void*)kp, nullptr, M, N, K);
        gemm_bf16<3><<<(waves + 7) / 8, 256, 0, stream>>>(cb, WvT, (void*)vp, nullptr, M, N, K);
    }
    // fused attention: one wave per (b, h, 16-row tile)
    {
        int blocks = B_ * HEADS * (N_ / 16);
        attn_kernel<<<blocks, 32, 0, stream>>>(qp, kp, vp, obuf);
    }
    // output projection + bias: [65536 x 320] @ Wo + bo -> f32 out
    {
        int M = B_ * N_, N = DQ, K = INNER;
        int waves = (M >> 4) * (N >> 6);
        gemm_bf16<0><<<(waves + 7) / 8, 256, 0, stream>>>(obuf, WoT, (void*)out, bo, M, N, K);
    }
    (void)in_sizes; (void)n_in; (void)out_size; (void)ws_size;
}
